// BaselineAttention_29764123361370
// MI455X (gfx1250) — compile-verified, matrix operand order checked by reading, not run on hardware
//
#include <hip/hip_runtime.h>
#include <hip/hip_bf16.h>

// ---------------------------------------------------------------------------
// BaselineAttention on MI455X (gfx1250): bf16 WMMA GEMMs + flash attention,
// with CDNA5 async global->LDS tile staging (ASYNCcnt) and double-buffered
// LDS pipelines.  B=4, S=2048, D=2048, H=16, HD=128.
// Workspace layout (bytes):
//   [  0MB) Xb    bf16 [8192,2048]          32MB
//   [ 32MB) Wqb   bf16 [2048,2048]           8MB
//   [ 40MB) Wkb                               8MB
//   [ 48MB) Wvb                               8MB
//   [ 56MB) Wob                               8MB
//   [ 64MB) Qb    bf16 [B,H,S,HD]           32MB
//   [ 96MB) Kb                               32MB
//   [128MB) Vb                               32MB
//   [160MB) Ob    bf16 [B,S,D]              32MB
// total 192MB.
// ---------------------------------------------------------------------------

typedef __bf16 v16bf __attribute__((ext_vector_type(16)));
typedef float v8f __attribute__((ext_vector_type(8)));
typedef unsigned int u32x8 __attribute__((ext_vector_type(8)));

__device__ __forceinline__ unsigned short f32_to_bf16(float f) {
  unsigned int u = __builtin_bit_cast(unsigned int, f);
  u += 0x7FFFu + ((u >> 16) & 1u);   // round-to-nearest-even
  return (unsigned short)(u >> 16);
}

__device__ __forceinline__ v16bf frag_cast(u32x8 t) {
  return __builtin_bit_cast(v16bf, t);
}

// CDNA5 async global->LDS copy, 16B per lane.  LDS dest offset is the low
// 32 bits of the generic address of a __shared__ object (LDS aperture maps
// addr[31:0] to the LDS offset).  Tracked by ASYNCcnt.
__device__ __forceinline__ void async_b128(void* lds_ptr, const void* gptr) {
  unsigned lds = (unsigned)(size_t)lds_ptr;
  asm volatile("global_load_async_to_lds_b128 %0, %1, off"
               :: "v"(lds), "v"(gptr)
               : "memory");
}
__device__ __forceinline__ void wait_async0() {
  asm volatile("s_wait_asynccnt 0" ::: "memory");
}

// fp32 -> bf16 elementwise, 4 elements/thread (n is a multiple of 4).
__global__ __launch_bounds__(256) void cvt_f32_bf16(
    const float* __restrict__ in, unsigned short* __restrict__ out, int n) {
  int i = (blockIdx.x * blockDim.x + threadIdx.x) * 4;
  if (i + 3 < n) {
    float4 f = *(const float4*)(in + i);
    unsigned long long p =
        (unsigned long long)f32_to_bf16(f.x) |
        ((unsigned long long)f32_to_bf16(f.y) << 16) |
        ((unsigned long long)f32_to_bf16(f.z) << 32) |
        ((unsigned long long)f32_to_bf16(f.w) << 48);
    *(unsigned long long*)(out + i) = p;
  }
}

// ---------------------------------------------------------------------------
// GEMM: out[M=8192, N=2048] = A[M,2048] * W[2048,N] + bias[N]
// Block tile 128x128, 256 threads (8 waves), wave tile 32x64, K-step 32.
// Double-buffered LDS tiles filled by async global->LDS copies.
// mode 0: write bf16 to outb in [B,H,S,HD] (head-split) layout.
// mode 1: write f32  to outf in [B,S,D] layout.
// ---------------------------------------------------------------------------
__global__ __launch_bounds__(256, 1) void gemm128(
    const unsigned short* __restrict__ A,
    const unsigned short* __restrict__ W,
    const float* __restrict__ bias,
    unsigned short* __restrict__ outb,
    float* __restrict__ outf,
    int mode) {
  constexpr int D = 2048;
  const int tid = threadIdx.x;
  const int lane = tid & 31;
  const int wid = tid >> 5;           // 0..7
  const int half = lane >> 4;
  const int l15 = lane & 15;
  const int wm = (wid >> 1) * 32;     // wave row offset in block tile
  const int wn = (wid & 1) * 64;      // wave col offset in block tile
  const int m0 = blockIdx.x * 128;
  const int n0 = blockIdx.y * 128;

  // +8 element pad: 16B-aligned rows for async B128, conflict-free gathers.
  __shared__ __align__(16) unsigned short Xs[2][128][40];
  __shared__ __align__(16) unsigned short Ws[2][32][136];

  // ISA 16-bit A-fragment K map: VGPR i holds K pair {k0,k0+1},
  // k0 = 2*(i&3) + 16*(i>>2) + 8*half.
  int kmap[8];
#pragma unroll
  for (int i = 0; i < 8; ++i)
    kmap[i] = ((i & 3) * 2) + ((i >> 2) * 16) + half * 8;

  v8f acc[2][4];
#pragma unroll
  for (int mi = 0; mi < 2; ++mi)
#pragma unroll
    for (int ni = 0; ni < 4; ++ni)
#pragma unroll
      for (int r = 0; r < 8; ++r) acc[mi][ni][r] = 0.0f;

  // stage one K-tile (A: 128x32, W: 32x128) into LDS buffer `buf`
  auto stage = [&](int kt, int buf) {
#pragma unroll
    for (int j = 0; j < 2; ++j) {
      int chunk = tid * 2 + j;               // 0..511
      int row = chunk >> 2;                  // 0..127
      int c8 = (chunk & 3) * 8;              // 0,8,16,24
      async_b128(&Xs[buf][row][c8],
                 A + (size_t)(m0 + row) * D + kt * 32 + c8);
    }
#pragma unroll
    for (int j = 0; j < 2; ++j) {
      int chunk = tid * 2 + j;
      int row = chunk >> 4;                  // 0..31
      int c8 = (chunk & 15) * 8;             // 0..120
      async_b128(&Ws[buf][row][c8],
                 W + (size_t)(kt * 32 + row) * D + n0 + c8);
    }
  };

  stage(0, 0);
  wait_async0();
  __syncthreads();

  for (int kt = 0; kt < D / 32; ++kt) {
    const int p = kt & 1;
    // overlap: fetch next tile into the other buffer while computing
    if (kt + 1 < D / 32) stage(kt + 1, 1 - p);

    // prefetch the K-tile after next into GL2 (speculative)
    if (kt + 2 < D / 32)
      __builtin_prefetch(A + (size_t)(m0 + (tid >> 1)) * D + (kt + 2) * 32, 0, 0);

    // --- fragments from buffer p ---
    v16bf afrag[2];
#pragma unroll
    for (int mi = 0; mi < 2; ++mi) {
      u32x8 t;
#pragma unroll
      for (int i = 0; i < 8; ++i)
        t[i] = *(const unsigned int*)&Xs[p][wm + mi * 16 + l15][kmap[i]];
      afrag[mi] = frag_cast(t);
    }
    v16bf bfrag[4];
#pragma unroll
    for (int ni = 0; ni < 4; ++ni) {
      int col = wn + ni * 16 + l15;
      u32x8 t;
#pragma unroll
      for (int i = 0; i < 8; ++i) {
        unsigned int lo = Ws[p][kmap[i]][col];
        unsigned int hi = Ws[p][kmap[i] + 1][col];
        t[i] = lo | (hi << 16);
      }
      bfrag[ni] = frag_cast(t);
    }
#pragma unroll
    for (int mi = 0; mi < 2; ++mi)
#pragma unroll
      for (int ni = 0; ni < 4; ++ni)
        acc[mi][ni] = __builtin_amdgcn_wmma_f32_16x16x32_bf16(
            false, afrag[mi], false, bfrag[ni], (short)0, acc[mi][ni],
            false, false);

    wait_async0();
    __syncthreads();
  }

  // --- epilogue: bias + store (C layout: lane=N, VGPR r = row r + 8*half) ---
#pragma unroll
  for (int mi = 0; mi < 2; ++mi)
#pragma unroll
    for (int ni = 0; ni < 4; ++ni) {
      int col = n0 + wn + ni * 16 + l15;
      float bv = bias[col];
#pragma unroll
      for (int r = 0; r < 8; ++r) {
        int row = m0 + wm + mi * 16 + r + half * 8;
        float v = acc[mi][ni][r] + bv;
        if (mode == 0) {
          int b = row >> 11, s = row & 2047;   // row = b*S + s
          int h = col >> 7, hd = col & 127;    // col = h*HD + hd
          outb[(((size_t)(b * 16 + h) * 2048) + s) * 128 + hd] = f32_to_bf16(v);
        } else {
          outf[(size_t)row * 2048 + col] = v;
        }
      }
    }
}

// ---------------------------------------------------------------------------
// Flash attention: one block = 4 waves x 16 query rows = 64 rows of one
// (b,h).  K/V staged as double-buffered 32x128 LDS tiles via async copies;
// online softmax in registers; P converted C-layout -> A-layout through
// per-wave LDS; O accumulated with P x V WMMAs.
// ---------------------------------------------------------------------------
__global__ __launch_bounds__(128, 1) void flash_attn(
    const unsigned short* __restrict__ Q,
    const unsigned short* __restrict__ K,
    const unsigned short* __restrict__ V,
    unsigned short* __restrict__ O) {
  constexpr int S = 2048, HD = 128, H = 16;
  const float scale = 0.08838834764831845f;  // 1/sqrt(128)
  const int tid = threadIdx.x;
  const int lane = tid & 31;
  const int w = tid >> 5;               // wave 0..3
  const int half = lane >> 4;
  const int l15 = lane & 15;
  const int qt = blockIdx.x & 31;       // S/64 query tiles
  const int bh = blockIdx.x >> 5;       // b*H + h
  const size_t hoff = (size_t)bh * S * HD;
  const int q0 = qt * 64 + w * 16;

  int kmap[8];
#pragma unroll
  for (int i = 0; i < 8; ++i)
    kmap[i] = ((i & 3) * 2) + ((i >> 2) * 16) + half * 8;

  // Q fragments for all 4 K-chunks of HD (contiguous 4B gathers from global).
  v16bf qf[4];
#pragma unroll
  for (int kc = 0; kc < 4; ++kc) {
    u32x8 t;
#pragma unroll
    for (int i = 0; i < 8; ++i)
      t[i] = *(const unsigned int*)(Q + hoff + (size_t)(q0 + l15) * HD +
                                    kc * 32 + kmap[i]);
    qf[kc] = frag_cast(t);
  }

  __shared__ __align__(16) unsigned short Kt[2][32][136];
  __shared__ __align__(16) unsigned short Vt[2][32][136];
  __shared__ unsigned short Ps[4][16][36];

  float mrow[8], lrow[8];
  v8f oac[8];
#pragma unroll
  for (int r = 0; r < 8; ++r) { mrow[r] = -3.0e38f; lrow[r] = 0.0f; }
#pragma unroll
  for (int t = 0; t < 8; ++t)
#pragma unroll
    for (int r = 0; r < 8; ++r) oac[t][r] = 0.0f;

  // stage K/V tiles (32x128 each; 512 x 16B chunks, 128 threads)
  auto stageKV = [&](int kt, int buf) {
#pragma unroll
    for (int j = 0; j < 4; ++j) {
      int chunk = tid * 4 + j;             // 0..511
      int row = chunk >> 4;                // 0..31
      int c8 = (chunk & 15) * 8;           // 0..120
      async_b128(&Kt[buf][row][c8],
                 K + hoff + (size_t)(kt * 32 + row) * HD + c8);
      async_b128(&Vt[buf][row][c8],
                 V + hoff + (size_t)(kt * 32 + row) * HD + c8);
    }
  };

  stageKV(0, 0);
  wait_async0();
  __syncthreads();

  for (int kt = 0; kt < S / 32; ++kt) {
    const int p = kt & 1;
    if (kt + 1 < S / 32) stageKV(kt + 1, 1 - p);

    // --- scores: S16x32 = Q(16x128) . K^T, two 16-wide key subtiles ---
    v8f sac[2];
#pragma unroll
    for (int ns = 0; ns < 2; ++ns) {
#pragma unroll
      for (int r = 0; r < 8; ++r) sac[ns][r] = 0.0f;
#pragma unroll
      for (int kc = 0; kc < 4; ++kc) {
        u32x8 t;
#pragma unroll
        for (int i = 0; i < 8; ++i)
          t[i] = *(const unsigned int*)&Kt[p][ns * 16 + l15][kc * 32 + kmap[i]];
        sac[ns] = __builtin_amdgcn_wmma_f32_16x16x32_bf16(
            false, qf[kc], false, frag_cast(t), (short)0, sac[ns],
            false, false);
      }
    }

    // --- online softmax (rows live across the 16 lanes of each half) ---
#pragma unroll
    for (int r = 0; r < 8; ++r) {
      float s0 = sac[0][r] * scale;
      float s1 = sac[1][r] * scale;
      float mx = fmaxf(s0, s1);
#pragma unroll
      for (int off = 1; off < 16; off <<= 1)
        mx = fmaxf(mx, __shfl_xor(mx, off, 32));
      float mnew = fmaxf(mrow[r], mx);
      float corr = __expf(mrow[r] - mnew);
      float p0 = __expf(s0 - mnew);
      float p1 = __expf(s1 - mnew);
      float ps = p0 + p1;
#pragma unroll
      for (int off = 1; off < 16; off <<= 1)
        ps += __shfl_xor(ps, off, 32);
      lrow[r] = lrow[r] * corr + ps;
      mrow[r] = mnew;
#pragma unroll
      for (int t = 0; t < 8; ++t) oac[t][r] *= corr;
      int prow = r + half * 8;
      Ps[w][prow][l15] = f32_to_bf16(p0);
      Ps[w][prow][16 + l15] = f32_to_bf16(p1);
    }

    // --- O += P(16x32) . V(32x128) ---
    u32x8 tp;
#pragma unroll
    for (int i = 0; i < 8; ++i)
      tp[i] = *(const unsigned int*)&Ps[w][l15][kmap[i]];
    v16bf pf = frag_cast(tp);
#pragma unroll
    for (int t = 0; t < 8; ++t) {
      u32x8 tv;
#pragma unroll
      for (int i = 0; i < 8; ++i) {
        unsigned int lo = Vt[p][kmap[i]][t * 16 + l15];
        unsigned int hi = Vt[p][kmap[i] + 1][t * 16 + l15];
        tv[i] = lo | (hi << 16);
      }
      oac[t] = __builtin_amdgcn_wmma_f32_16x16x32_bf16(
          false, pf, false, frag_cast(tv), (short)0, oac[t], false, false);
    }

    wait_async0();
    __syncthreads();
  }

  // --- epilogue: O / l, write bf16 to [B,S,D] merged-head layout ---
  const int b = bh >> 4, h = bh & 15;
#pragma unroll
  for (int r = 0; r < 8; ++r) {
    float inv = 1.0f / lrow[r];
    int s = q0 + r + half * 8;
    size_t base = (((size_t)(b * S + s)) * H + h) * (size_t)HD;
#pragma unroll
    for (int t = 0; t < 8; ++t)
      O[base + t * 16 + l15] = f32_to_bf16(oac[t][r] * inv);
  }
  __builtin_amdgcn_s_wait_tensorcnt(0);
}

// ---------------------------------------------------------------------------
extern "C" void kernel_launch(void* const* d_in, const int* in_sizes, int n_in,
                              void* d_out, int out_size, void* d_ws,
                              size_t ws_size, hipStream_t stream) {
  (void)in_sizes; (void)n_in; (void)out_size; (void)ws_size;
  const float* x  = (const float*)d_in[0];
  const float* wq = (const float*)d_in[1];
  const float* bq = (const float*)d_in[2];
  const float* wk = (const float*)d_in[3];
  const float* bk = (const float*)d_in[4];
  const float* wv = (const float*)d_in[5];
  const float* bv = (const float*)d_in[6];
  const float* wo = (const float*)d_in[7];
  const float* bo = (const float*)d_in[8];
  float* out = (float*)d_out;

  char* ws = (char*)d_ws;
  const size_t MB = 1ull << 20;
  unsigned short* Xb  = (unsigned short*)(ws + 0 * MB);
  unsigned short* Wqb = (unsigned short*)(ws + 32 * MB);
  unsigned short* Wkb = (unsigned short*)(ws + 40 * MB);
  unsigned short* Wvb = (unsigned short*)(ws + 48 * MB);
  unsigned short* Wob = (unsigned short*)(ws + 56 * MB);
  unsigned short* Qb  = (unsigned short*)(ws + 64 * MB);
  unsigned short* Kb  = (unsigned short*)(ws + 96 * MB);
  unsigned short* Vb  = (unsigned short*)(ws + 128 * MB);
  unsigned short* Ob  = (unsigned short*)(ws + 160 * MB);

  const int nx = 8192 * 2048;        // X elements
  const int nw = 2048 * 2048;        // weight elements
  const int cb = 256;
  cvt_f32_bf16<<<(nx / 4 + cb - 1) / cb, cb, 0, stream>>>(x, Xb, nx);
  cvt_f32_bf16<<<(nw / 4 + cb - 1) / cb, cb, 0, stream>>>(wq, Wqb, nw);
  cvt_f32_bf16<<<(nw / 4 + cb - 1) / cb, cb, 0, stream>>>(wk, Wkb, nw);
  cvt_f32_bf16<<<(nw / 4 + cb - 1) / cb, cb, 0, stream>>>(wv, Wvb, nw);
  cvt_f32_bf16<<<(nw / 4 + cb - 1) / cb, cb, 0, stream>>>(wo, Wob, nw);

  dim3 gg(64, 16, 1), gb(256, 1, 1);
  gemm128<<<gg, gb, 0, stream>>>(Xb, Wqb, bq, Qb, nullptr, 0);
  gemm128<<<gg, gb, 0, stream>>>(Xb, Wkb, bk, Kb, nullptr, 0);
  gemm128<<<gg, gb, 0, stream>>>(Xb, Wvb, bv, Vb, nullptr, 0);

  flash_attn<<<2048, 128, 0, stream>>>(Qb, Kb, Vb, Ob);

  gemm128<<<gg, gb, 0, stream>>>(Ob, Wob, bo, nullptr, out, 1);
}
